// CustomLSTMCell_18021682774684
// MI455X (gfx1250) — matrix operand order repaired and checked by reading, not compiled, not run on hardware
//
#include <hip/hip_runtime.h>
#include <math.h>

typedef __attribute__((ext_vector_type(16))) __bf16 v16bf;
typedef __attribute__((ext_vector_type(8)))  float  v8f;
typedef unsigned short u16;
typedef unsigned int   u32;

// Problem dims (fixed by the reference)
constexpr int BDIM = 8192;
constexpr int DDIM = 512;
constexpr int UDIM = 512;
constexpr int N4U  = 2048;
constexpr int KTOT = 1024;   // D + U fused K

// GEMM tile: 256 threads = 8 wave32; 128 rows x (16 units x 4 gates); BK = 32
constexpr int BM = 128;
constexpr int BN = 16;
constexpr int BK = 32;
constexpr int LS = BK + 8;            // LDS row stride in u16 (80 B, 16B aligned)
constexpr int NIT = KTOT / BK;        // 32 k-steps

// LDS arena (ping-pong): [AsH | AsL | BsH | BsL]
constexpr int ASZ    = BM * LS;       // 5120 u16
constexpr int BSZ    = 4 * BN * LS;   // 2560 u16
constexpr int OFF_AH = 0;
constexpr int OFF_AL = ASZ;
constexpr int OFF_BH = 2 * ASZ;
constexpr int OFF_BL = 2 * ASZ + BSZ;
constexpr int TOT    = 2 * ASZ + 2 * BSZ;   // 15360 u16 per buffer

// Workspace layout (bytes)
constexpr size_t A_ELEMS = (size_t)BDIM * KTOT;
constexpr size_t B_ELEMS = (size_t)N4U  * KTOT;
constexpr size_t WS_AHI  = 0;
constexpr size_t WS_ALO  = A_ELEMS * 2;
constexpr size_t WS_BHI  = A_ELEMS * 4;
constexpr size_t WS_BLO  = A_ELEMS * 4 + B_ELEMS * 2;
constexpr size_t WS_NEED = A_ELEMS * 4 + B_ELEMS * 4;   // 40 MiB

__device__ __forceinline__ void bsplit(float x, u16& hi, u16& lo) {
    u32 u = __float_as_uint(x);
    hi = (u16)(u >> 16);
    float rem = x - __uint_as_float(u & 0xFFFF0000u);
    lo = (u16)(__float_as_uint(rem) >> 16);
}
__device__ __forceinline__ void bsplit2(float a, float b, u32& hp, u32& lp) {
    u16 ha, la, hb, lb;
    bsplit(a, ha, la); bsplit(b, hb, lb);
    hp = (u32)ha | ((u32)hb << 16);
    lp = (u32)la | ((u32)lb << 16);
}
// branch-free activations (safe at +-inf)
__device__ __forceinline__ float fast_sigmoid(float x) {
    return 1.0f / (1.0f + __expf(-x));
}
__device__ __forceinline__ float fast_tanh(float x) {
    return 2.0f / (1.0f + __expf(-2.0f * x)) - 1.0f;
}

// ---- Phase 1a: x|h -> bf16 hi/lo activations, [B][1024] ----
__global__ __launch_bounds__(256)
void conv_acts(const float* __restrict__ x, const float* __restrict__ h,
               u16* __restrict__ Ahi, u16* __restrict__ Alo) {
    size_t t = (size_t)blockIdx.x * 256 + threadIdx.x;
    int r  = (int)(t >> 8);
    int kq = (int)(t & 255) << 2;
    const float* src = (kq < DDIM) ? (x + (size_t)r * DDIM + kq)
                                   : (h + (size_t)r * UDIM + (kq - DDIM));
    float4 v = *reinterpret_cast<const float4*>(src);
    uint2 hp, lp;
    bsplit2(v.x, v.y, hp.x, lp.x);
    bsplit2(v.z, v.w, hp.y, lp.y);
    *reinterpret_cast<uint2*>(Ahi + (size_t)r * KTOT + kq) = hp;
    *reinterpret_cast<uint2*>(Alo + (size_t)r * KTOT + kq) = lp;
}

// ---- Phase 1b: W;R -> transposed bf16 hi/lo weights, [2048][1024] ----
__global__ __launch_bounds__(256)
void conv_wts(const float* __restrict__ W, const float* __restrict__ R,
              u16* __restrict__ Bhi, u16* __restrict__ Blo) {
    __shared__ float tile[16][17];
    const int k0 = blockIdx.x * 16;
    const int n0 = blockIdx.y * 16;
    const int tk = threadIdx.x >> 4;
    const int tn = threadIdx.x & 15;
    const float* src = (k0 < DDIM) ? W : R;
    const int kl = (k0 < DDIM) ? k0 : k0 - DDIM;
    tile[tk][tn] = src[(size_t)(kl + tk) * N4U + n0 + tn];
    __syncthreads();
    float v = tile[tn][tk];
    u16 hi, lo; bsplit(v, hi, lo);
    size_t o = (size_t)(n0 + tk) * KTOT + k0 + tn;
    Bhi[o] = hi; Blo[o] = lo;
}

// ---- Phase 2: fused GEMM + gate epilogue ----
template<bool PRE>
__global__ __launch_bounds__(256)
void lstm_gemm(const float* __restrict__ x,  const float* __restrict__ h,
               const float* __restrict__ cin, const float* __restrict__ W,
               const float* __restrict__ R,  const float* __restrict__ bias,
               const u16* __restrict__ Ahi, const u16* __restrict__ Alo,
               const u16* __restrict__ Bhi, const u16* __restrict__ Blo,
               float* __restrict__ out) {
    __shared__ __align__(16) u16 S[2 * TOT];

    const int tid  = threadIdx.x;
    const int wave = tid >> 5;
    const int lane = tid & 31;
    const int half = lane >> 4;
    const int l16  = lane & 15;
    const int row0 = blockIdx.x * BM;
    const int u0   = blockIdx.y * BN;

    v8f acc[4] = {};
    const int fr_a = (wave * 16 + l16) * LS + half * 8;   // + {0,16} chunks
    const int kb0  = half * 8;

    // shared compute step: fragments from `base`, 12 WMMAs
    auto compute = [&](const u16* base) {
        union FR { v16bf v; uint4 q[2]; };
        FR ah, al;
        ah.q[0] = *reinterpret_cast<const uint4*>(base + OFF_AH + fr_a);
        ah.q[1] = *reinterpret_cast<const uint4*>(base + OFF_AH + fr_a + 16);
        al.q[0] = *reinterpret_cast<const uint4*>(base + OFF_AL + fr_a);
        al.q[1] = *reinterpret_cast<const uint4*>(base + OFF_AL + fr_a + 16);
        #pragma unroll
        for (int g = 0; g < 4; ++g) {
            const int brow = (g * BN + l16) * LS + kb0;
            FR bh, bl;
            bh.q[0] = *reinterpret_cast<const uint4*>(base + OFF_BH + brow);
            bh.q[1] = *reinterpret_cast<const uint4*>(base + OFF_BH + brow + 16);
            bl.q[0] = *reinterpret_cast<const uint4*>(base + OFF_BL + brow);
            bl.q[1] = *reinterpret_cast<const uint4*>(base + OFF_BL + brow + 16);
            acc[g] = __builtin_amdgcn_wmma_f32_16x16x32_bf16(
                false, ah.v, false, bh.v, (short)0, acc[g], false, false);
            acc[g] = __builtin_amdgcn_wmma_f32_16x16x32_bf16(
                false, al.v, false, bh.v, (short)0, acc[g], false, false);
            acc[g] = __builtin_amdgcn_wmma_f32_16x16x32_bf16(
                false, ah.v, false, bl.v, (short)0, acc[g], false, false);
        }
    };

    if constexpr (PRE) {
        // k-invariant per-thread copy descriptors (6 x 16B per thread per step)
        u32        a_lds[4]; const u16* a_glb[4];
        u32        b_lds[2]; const u16* b_glb[2];
        const u32 lds0 = (u32)(size_t)(const void*)S;   // low 32b = LDS address
        #pragma unroll
        for (int i = 0; i < 4; ++i) {
            int cidx = i * 256 + tid, sel = cidx >> 9, cc = cidx & 511;
            int r = cc >> 2, ko = (cc & 3) * 8;
            a_lds[i] = lds0 + (u32)(((sel ? OFF_AL : OFF_AH) + r * LS + ko) * 2);
            a_glb[i] = (sel ? Alo : Ahi) + (size_t)(row0 + r) * KTOT + ko;
        }
        #pragma unroll
        for (int i = 0; i < 2; ++i) {
            int cidx = i * 256 + tid, sel = cidx >> 8, cc = cidx & 255;
            int lr = cc >> 2, ko = (cc & 3) * 8;
            int g = lr >> 4, n = lr & 15;
            b_lds[i] = lds0 + (u32)(((sel ? OFF_BL : OFF_BH) + lr * LS + ko) * 2);
            b_glb[i] = (sel ? Blo : Bhi) + (size_t)(g * UDIM + u0 + n) * KTOT + ko;
        }

        // async DMA global -> LDS (no data VGPRs, tracked by ASYNCcnt)
        auto async_fetch = [&](int it) {
            const int k0   = it * BK;
            const u32 bufo = (u32)((it & 1) * TOT * 2);
            #pragma unroll
            for (int i = 0; i < 4; ++i) {
                u32 l = a_lds[i] + bufo;
                const void* g = (const void*)(a_glb[i] + k0);
                asm volatile("global_load_async_to_lds_b128 %0, %1, off"
                             :: "v"(l), "v"(g) : "memory");
            }
            #pragma unroll
            for (int i = 0; i < 2; ++i) {
                u32 l = b_lds[i] + bufo;
                const void* g = (const void*)(b_glb[i] + k0);
                asm volatile("global_load_async_to_lds_b128 %0, %1, off"
                             :: "v"(l), "v"(g) : "memory");
            }
        };

        async_fetch(0);
        asm volatile("s_wait_asynccnt 0x0" ::: "memory");
        __syncthreads();
        for (int it = 0; it < NIT; ++it) {
            if (it + 1 < NIT) async_fetch(it + 1);   // DMA fills other buffer
            compute(S + (it & 1) * TOT);
            if (it + 1 < NIT) {
                asm volatile("s_wait_asynccnt 0x0" ::: "memory");
                __syncthreads();
            }
        }
    } else {
        // fallback: single buffer, fetch->convert->store fused (short live ranges)
        for (int it = 0; it < NIT; ++it) {
            const int k0 = it * BK;
            const bool fh = (k0 < DDIM);
            const float* aB = fh ? x : h;
            const float* wB = fh ? W : R;
            const int kl = fh ? k0 : k0 - DDIM;
            __syncthreads();
            #pragma unroll
            for (int i = 0; i < 4; ++i) {
                int q = i * 256 + tid;
                int r = q >> 3, ko = (q & 7) * 4;
                float4 v = *reinterpret_cast<const float4*>(
                    aB + (size_t)(row0 + r) * DDIM + kl + ko);
                u32 h01, l01, h23, l23;
                bsplit2(v.x, v.y, h01, l01);
                bsplit2(v.z, v.w, h23, l23);
                int o = r * LS + ko;
                *reinterpret_cast<u32*>(S + OFF_AH + o)     = h01;
                *reinterpret_cast<u32*>(S + OFF_AH + o + 2) = h23;
                *reinterpret_cast<u32*>(S + OFF_AL + o)     = l01;
                *reinterpret_cast<u32*>(S + OFF_AL + o + 2) = l23;
            }
            #pragma unroll
            for (int i = 0; i < 8; ++i) {
                int e = i * 256 + tid;
                int g = e >> 9, kk = (e & 511) >> 4, n = e & 15;
                float v = wB[(size_t)(kl + kk) * N4U + g * UDIM + u0 + n];
                u16 hi, lo; bsplit(v, hi, lo);
                int o = (g * BN + n) * LS + kk;
                S[OFF_BH + o] = hi;
                S[OFF_BL + o] = lo;
            }
            __syncthreads();
            compute(S);
        }
    }

    // ---- Fused gate epilogue (C layout: VGPR vi -> M = vi + 8*half, N = l16) ----
    const int u = u0 + l16;
    const float bi = bias[u];
    const float bf = bias[UDIM + u];
    const float bg = bias[2 * UDIM + u];
    const float bo = bias[3 * UDIM + u];
    const size_t BU = (size_t)BDIM * UDIM;

    #pragma unroll
    for (int vi = 0; vi < 8; ++vi) {
        const int row = row0 + wave * 16 + half * 8 + vi;
        const size_t idx = (size_t)row * UDIM + u;
        float ig = fast_sigmoid(acc[0][vi] + bi);
        float fg = fast_sigmoid(acc[1][vi] + bf);
        float gg = fast_tanh(acc[2][vi] + bg);
        float og = fast_sigmoid(acc[3][vi] + bo);
        float c = fg * cin[idx] + ig * gg;
        float hh = og * fast_tanh(c);
        out[idx]          = hh;
        out[BU + idx]     = hh;
        out[2 * BU + idx] = c;
    }
}

extern "C" void kernel_launch(void* const* d_in, const int* in_sizes, int n_in,
                              void* d_out, int out_size, void* d_ws, size_t ws_size,
                              hipStream_t stream) {
    (void)in_sizes; (void)n_in; (void)out_size;
    const float* xin  = (const float*)d_in[0];
    const float* hin  = (const float*)d_in[1];
    const float* cin  = (const float*)d_in[2];
    const float* Wk   = (const float*)d_in[3];
    const float* Rk   = (const float*)d_in[4];
    const float* bias = (const float*)d_in[5];
    float* out = (float*)d_out;

    dim3 gemm_grid(BDIM / BM, UDIM / BN);   // 64 x 32
    dim3 gemm_block(256);

    if (d_ws != nullptr && ws_size >= WS_NEED) {
        char* ws = (char*)d_ws;
        u16* Ahi = (u16*)(ws + WS_AHI);
        u16* Alo = (u16*)(ws + WS_ALO);
        u16* Bhi = (u16*)(ws + WS_BHI);
        u16* Blo = (u16*)(ws + WS_BLO);
        conv_acts<<<(unsigned)(A_ELEMS / 4 / 256), 256, 0, stream>>>(xin, hin, Ahi, Alo);
        conv_wts<<<dim3(KTOT / 16, N4U / 16), 256, 0, stream>>>(Wk, Rk, Bhi, Blo);
        lstm_gemm<true><<<gemm_grid, gemm_block, 0, stream>>>(
            xin, hin, cin, Wk, Rk, bias, Ahi, Alo, Bhi, Blo, out);
    } else {
        lstm_gemm<false><<<gemm_grid, gemm_block, 0, stream>>>(
            xin, hin, cin, Wk, Rk, bias, nullptr, nullptr, nullptr, nullptr, out);
    }
}
